// CaptioningRNN_44092134260749
// MI455X (gfx1250) — matrix operand order repaired and checked by reading, not compiled, not run on hardware
//
#include <hip/hip_runtime.h>
#include <hip/hip_bf16.h>
#include <math.h>

typedef __bf16 bf16_t;
typedef __attribute__((ext_vector_type(16))) __bf16 v16bf;
typedef __attribute__((ext_vector_type(8)))  __bf16 v8bf;
typedef __attribute__((ext_vector_type(8)))  float  v8f;

// Problem constants
#define NB   128      // batch
#define TT   32       // timesteps
#define DD   400      // feature depth
#define WW   512      // embed dim
#define HH   1024     // hidden
#define VV   10000    // vocab
#define H4   4096     // 4*H
#define KC   2560     // W + H + H  (concat K for z gemm)

#if defined(__has_builtin)
#if __has_builtin(__builtin_amdgcn_global_load_async_to_lds_b128) && \
    __has_builtin(__builtin_amdgcn_s_wait_asynccnt)
#define HAVE_ASYNC_LDS 1
#endif
#endif

#ifdef HAVE_ASYNC_LDS
typedef int i4vec __attribute__((__vector_size__(16)));
typedef __attribute__((address_space(1))) i4vec g_i4;
typedef __attribute__((address_space(3))) i4vec l_i4;
__device__ __forceinline__ void async_copy_b128(const void* gsrc, void* ldst) {
  g_i4* g = (g_i4*)(uintptr_t)gsrc;
  l_i4* l = (l_i4*)(unsigned)(uintptr_t)ldst;  // low 32 bits = LDS offset
  __builtin_amdgcn_global_load_async_to_lds_b128(g, l, 0, 0);
}
#endif

__device__ __forceinline__ float sigf(float x) { return 1.0f / (1.0f + __expf(-x)); }

// ---- WMMA fragment loaders (CDNA5 16-bit layouts, wave32) -------------------
// A 16x32 bf16: lane m = lane&15, kb = lane<16?0:8.
// elements 0..7  -> K = k0+kb+e ; elements 8..15 -> K = k0+kb+16+(e-8)
__device__ __forceinline__ v16bf load_a_frag(const bf16_t* tile, int ldk, int k0, int lane) {
  int m  = lane & 15;
  int kb = (lane < 16) ? 0 : 8;
  const v8bf* p0 = (const v8bf*)(tile + m * ldk + k0 + kb);
  const v8bf* p1 = (const v8bf*)(tile + m * ldk + k0 + kb + 16);
  v8bf lo = *p0, hi = *p1;
  return __builtin_shufflevector(lo, hi, 0,1,2,3,4,5,6,7,8,9,10,11,12,13,14,15);
}
// B 32x16 bf16 from a K-contiguous (transposed) weight row pointer.
// kb = lane<16?0:8; elements 0..7 -> K = k0+kb+e ; elements 8..15 -> K = k0+kb+16+(e-8)
__device__ __forceinline__ v16bf load_b_frag_p(const bf16_t* rowPtr, int k0, int lane) {
  int kb = (lane < 16) ? 0 : 8;
  v8bf lo = *(const v8bf*)(rowPtr + k0 + kb);
  v8bf hi = *(const v8bf*)(rowPtr + k0 + kb + 16);
  return __builtin_shufflevector(lo, hi, 0,1,2,3,4,5,6,7,8,9,10,11,12,13,14,15);
}

// ---------------- Prep kernels ----------------------------------------------
__global__ void k_wcat(const float* __restrict__ Wx, const float* __restrict__ Wh,
                       const float* __restrict__ Wattn, bf16_t* __restrict__ WcatT) {
  size_t idx = (size_t)blockIdx.x * blockDim.x + threadIdx.x;
  if (idx >= (size_t)KC * H4) return;
  int k = (int)(idx / H4);
  int j = (int)(idx % H4);
  float v;
  if (k < WW)            v = Wx[(size_t)k * H4 + j];
  else if (k < WW + HH)  v = Wh[(size_t)(k - WW) * H4 + j];
  else                   v = Wattn[(size_t)(k - WW - HH) * H4 + j];
  WcatT[(size_t)j * KC + k] = (bf16_t)v;
}

__global__ void k_wvocab(const float* __restrict__ Wv, bf16_t* __restrict__ WvT) {
  size_t idx = (size_t)blockIdx.x * blockDim.x + threadIdx.x;
  if (idx >= (size_t)HH * VV) return;
  int k = (int)(idx / VV);
  int v = (int)(idx % VV);
  WvT[(size_t)v * HH + k] = (bf16_t)Wv[idx];
}

__global__ void k_embed(const int* __restrict__ captions, const float* __restrict__ We,
                        bf16_t* __restrict__ x) {
  size_t idx = (size_t)blockIdx.x * blockDim.x + threadIdx.x;
  if (idx >= (size_t)TT * NB * WW) return;
  int w  = (int)(idx & (WW - 1));
  int nt = (int)(idx >> 9);
  int n  = nt & (NB - 1);
  int t  = nt >> 7;
  int tok = captions[n * 33 + t];
  x[idx] = (bf16_t)We[(size_t)tok * WW + w];
}

__global__ void __launch_bounds__(256) k_proj(const float* __restrict__ feat,
                                              const float* __restrict__ Wp,
                                              const float* __restrict__ bp,
                                              float* __restrict__ A_flat,
                                              float* __restrict__ hbuf,
                                              float* __restrict__ cbuf) {
  __shared__ float f[DD * 16];   // 25.6 KB
  int n = blockIdx.x, tid = threadIdx.x;
  for (int i = tid; i < DD * 16; i += 256) f[i] = feat[(size_t)n * DD * 16 + i];
  __syncthreads();
  for (int q = 0; q < 4; q++) {
    int h = q * 256 + tid;
    float acc[16];
    #pragma unroll
    for (int l = 0; l < 16; l++) acc[l] = 0.f;
    for (int d = 0; d < DD; d++) {
      float wp = Wp[(size_t)d * HH + h];
      const float* fr = f + d * 16;
      #pragma unroll
      for (int l = 0; l < 16; l++) acc[l] = fmaf(fr[l], wp, acc[l]);
    }
    float bias = bp[h], s = 0.f;
    float* ar = A_flat + ((size_t)n * HH + h) * 16;
    #pragma unroll
    for (int l = 0; l < 16; l++) { float a = acc[l] + bias; ar[l] = a; s += a; }
    float h0 = s * (1.0f / 16.0f);
    hbuf[n * HH + h] = h0;
    cbuf[n * HH + h] = h0;
  }
}

// ---------------- Per-timestep kernels --------------------------------------
__global__ void __launch_bounds__(256) k_attend(int t, const float* __restrict__ hbuf,
                                                const float* __restrict__ A_flat,
                                                const bf16_t* __restrict__ xall,
                                                bf16_t* __restrict__ XHA) {
  __shared__ float red[256 * 16];   // 16 KB
  int n = blockIdx.x, tid = threadIdx.x;
  float part[16];
  #pragma unroll
  for (int l = 0; l < 16; l++) part[l] = 0.f;
  for (int h = tid; h < HH; h += 256) {
    float ph = hbuf[n * HH + h];
    const float* ar = A_flat + ((size_t)n * HH + h) * 16;
    #pragma unroll
    for (int l = 0; l < 16; l++) part[l] = fmaf(ph, ar[l], part[l]);
  }
  #pragma unroll
  for (int l = 0; l < 16; l++) red[tid * 16 + l] = part[l];
  __syncthreads();
  for (int s = 128; s >= 1; s >>= 1) {
    if (tid < s)
      #pragma unroll
      for (int l = 0; l < 16; l++) red[tid * 16 + l] += red[(tid + s) * 16 + l];
    __syncthreads();
  }
  float e[16], m = -1e30f;
  #pragma unroll
  for (int l = 0; l < 16; l++) { e[l] = red[l] * (1.0f / 32.0f); m = fmaxf(m, e[l]); }
  float den = 0.f;
  #pragma unroll
  for (int l = 0; l < 16; l++) { e[l] = __expf(e[l] - m); den += e[l]; }
  float inv = 1.0f / den;
  #pragma unroll
  for (int l = 0; l < 16; l++) e[l] *= inv;

  bf16_t* row = XHA + (size_t)n * KC;
  for (int i = tid; i < WW; i += 256) row[i] = xall[((size_t)t * NB + n) * WW + i];
  for (int h = tid; h < HH; h += 256) {
    float ph = hbuf[n * HH + h];
    const float* ar = A_flat + ((size_t)n * HH + h) * 16;
    float at = 0.f;
    #pragma unroll
    for (int l = 0; l < 16; l++) at = fmaf(ar[l], e[l], at);
    row[WW + h]      = (bf16_t)ph;
    row[WW + HH + h] = (bf16_t)at;
  }
}

// z(128,4096) = XHA(128,2560) @ Wcat(2560,4096) + b
// 2-column-tile register blocking: one A-frag feeds two WMMAs.
__global__ void __launch_bounds__(256) k_zgemm(const bf16_t* __restrict__ XHA,
                                               const bf16_t* __restrict__ WcatT,
                                               const float* __restrict__ bias,
                                               float* __restrict__ z) {
  __shared__ bf16_t atile[16 * 512];   // 16 KB K-chunk of 16 rows
  int tid  = threadIdx.x;
  int wave = tid >> 5, lane = tid & 31;
  int rowBase = blockIdx.y * 16;
  int col0 = blockIdx.x * 256 + wave * 32 + (lane & 15);
  int col1 = col0 + 16;
  const bf16_t* bp0 = WcatT + (size_t)col0 * KC;
  const bf16_t* bp1 = WcatT + (size_t)col1 * KC;
  v8f acc0 = {}, acc1 = {};
  for (int c = 0; c < KC / 512; c++) {
    for (int idx = tid; idx < 1024; idx += 256) {   // 16 rows x 512 bf16
      int i = idx >> 6, jj = idx & 63;
      ((uint4*)atile)[idx] =
          ((const uint4*)(XHA + (size_t)(rowBase + i) * KC + c * 512))[jj];
    }
    __syncthreads();
    #pragma unroll
    for (int kk = 0; kk < 16; kk++) {
      int kg = c * 512 + kk * 32;
      v16bf a  = load_a_frag(atile, 512, kk * 32, lane);
      v16bf b0 = load_b_frag_p(bp0, kg, lane);
      v16bf b1 = load_b_frag_p(bp1, kg, lane);
      acc0 = __builtin_amdgcn_wmma_f32_16x16x32_bf16(false, a, false, b0, (short)0, acc0, false, false);
      acc1 = __builtin_amdgcn_wmma_f32_16x16x32_bf16(false, a, false, b1, (short)0, acc1, false, false);
    }
    __syncthreads();
  }
  float bv0 = bias[col0], bv1 = bias[col1];
  #pragma unroll
  for (int r = 0; r < 8; r++) {
    int m = r + ((lane < 16) ? 0 : 8);
    z[(size_t)(rowBase + m) * H4 + col0] = acc0[r] + bv0;
    z[(size_t)(rowBase + m) * H4 + col1] = acc1[r] + bv1;
  }
}

__global__ void k_lstm(int t, const float* __restrict__ z, float* __restrict__ hbuf,
                       float* __restrict__ cbuf, bf16_t* __restrict__ hn) {
  int idx = blockIdx.x * blockDim.x + threadIdx.x;
  if (idx >= NB * HH) return;
  int n = idx >> 10, h = idx & (HH - 1);
  const float* zr = z + (size_t)n * H4;
  float ai = zr[h], af = zr[HH + h], ao = zr[2 * HH + h], ag = zr[3 * HH + h];
  float c  = sigf(af) * cbuf[idx] + sigf(ai) * tanhf(ag);
  float hh = sigf(ao) * tanhf(c);
  cbuf[idx] = c;
  hbuf[idx] = hh;
  hn[((size_t)t * NB + n) * HH + h] = (bf16_t)hh;
}

// ---------------- Vocab head: WMMA GEMM + per-lane online log-sum-exp -------
__global__ void __launch_bounds__(256) k_vocab(const bf16_t* __restrict__ hn,
                                               const bf16_t* __restrict__ WvT,
                                               const float* __restrict__ bvoc,
                                               const int* __restrict__ captions,
                                               float* __restrict__ nllbuf) {
  __shared__ bf16_t atile[16 * HH];            // 32 KB: 16 rows of h (bf16)
  __shared__ float red_m[8][16], red_s[8][16], red_t[8][16];
  int tid = threadIdx.x, wave = tid >> 5, lane = tid & 31;
  int rowBase = blockIdx.x * 16;
  // Stage 16x1024 bf16 = 2048 x b128 transfers
#ifdef HAVE_ASYNC_LDS
  for (int idx = tid; idx < 2048; idx += 256) {
    int i = idx >> 7, jj = idx & 127;
    async_copy_b128(hn + (size_t)(rowBase + i) * HH + jj * 8, &atile[i * HH + jj * 8]);
  }
  __builtin_amdgcn_s_wait_asynccnt(0);
  __syncthreads();
#else
  for (int idx = tid; idx < 2048; idx += 256) {
    int i = idx >> 7, jj = idx & 127;
    ((uint4*)atile)[idx] = ((const uint4*)(hn + (size_t)(rowBase + i) * HH))[jj];
  }
  __syncthreads();
#endif

  int half = (lane < 16) ? 0 : 8;
  int target[8];
  #pragma unroll
  for (int r = 0; r < 8; r++) {
    int rg = rowBase + r + half;
    int t = rg >> 7, n = rg & 127;
    target[r] = captions[n * 33 + t + 1];   // cap_out
  }
  // Per-lane running log-sum-exp over this lane's columns (no shfl in loop)
  float m_run[8], s_run[8], tgt[8];
  #pragma unroll
  for (int r = 0; r < 8; r++) { m_run[r] = -1e30f; s_run[r] = 0.f; tgt[r] = 0.f; }

  for (int vt0 = wave * 2; vt0 < VV / 16; vt0 += 16) {
    bool has1 = (vt0 + 1) < (VV / 16);
    int v0 = vt0 * 16 + (lane & 15);
    int v1 = v0 + 16;
    const bf16_t* bp0 = WvT + (size_t)v0 * HH;
    const bf16_t* bp1 = WvT + (size_t)(has1 ? v1 : v0) * HH;
    if (vt0 + 16 < VV / 16)
      __builtin_prefetch(WvT + (size_t)(v0 + 256) * HH, 0, 0);
    v8f acc0 = {}, acc1 = {};
    #pragma unroll
    for (int kk = 0; kk < 32; kk++) {
      v16bf a  = load_a_frag(atile, HH, kk * 32, lane);
      v16bf b0 = load_b_frag_p(bp0, kk * 32, lane);
      v16bf b1 = load_b_frag_p(bp1, kk * 32, lane);
      acc0 = __builtin_amdgcn_wmma_f32_16x16x32_bf16(false, a, false, b0, (short)0, acc0, false, false);
      acc1 = __builtin_amdgcn_wmma_f32_16x16x32_bf16(false, a, false, b1, (short)0, acc1, false, false);
    }
    float bv0 = bvoc[v0];
    #pragma unroll
    for (int r = 0; r < 8; r++) {
      float sv = acc0[r] + bv0;
      float mn = fmaxf(m_run[r], sv);
      s_run[r] = s_run[r] * __expf(m_run[r] - mn) + __expf(sv - mn);
      m_run[r] = mn;
      if (v0 == target[r]) tgt[r] += sv;
    }
    if (has1) {
      float bv1 = bvoc[v1];
      #pragma unroll
      for (int r = 0; r < 8; r++) {
        float sv = acc1[r] + bv1;
        float mn = fmaxf(m_run[r], sv);
        s_run[r] = s_run[r] * __expf(m_run[r] - mn) + __expf(sv - mn);
        m_run[r] = mn;
        if (v1 == target[r]) tgt[r] += sv;
      }
    }
  }
  // Merge the 16 lanes of each half (pairwise LSE merge)
  #pragma unroll
  for (int mask = 1; mask < 16; mask <<= 1) {
    #pragma unroll
    for (int r = 0; r < 8; r++) {
      float mo = __shfl_xor(m_run[r], mask, 32);
      float so = __shfl_xor(s_run[r], mask, 32);
      float M  = fmaxf(m_run[r], mo);
      s_run[r] = s_run[r] * __expf(m_run[r] - M) + so * __expf(mo - M);
      m_run[r] = M;
      tgt[r]  += __shfl_xor(tgt[r], mask, 32);
    }
  }
  if ((lane & 15) == 0) {
    #pragma unroll
    for (int r = 0; r < 8; r++) {
      int rw = r + half;
      red_m[wave][rw] = m_run[r];
      red_s[wave][rw] = s_run[r];
      red_t[wave][rw] = tgt[r];
    }
  }
  __syncthreads();
  if (tid < 16) {
    int row = tid, rg = rowBase + row;
    float M = -1e30f;
    for (int w = 0; w < 8; w++) M = fmaxf(M, red_m[w][row]);
    float S = 0.f, Tv = 0.f;
    for (int w = 0; w < 8; w++) {
      S  += red_s[w][row] * __expf(red_m[w][row] - M);
      Tv += red_t[w][row];
    }
    int t = rg >> 7, n = rg & 127;
    int tg = captions[n * 33 + t + 1];
    nllbuf[rg] = (tg != 0) ? ((M + logf(S)) - Tv) : 0.f;
  }
}

__global__ void __launch_bounds__(256) k_loss(const float* __restrict__ nllbuf, float* __restrict__ out) {
  __shared__ float red[256];
  int tid = threadIdx.x;
  float s = 0.f;
  for (int i = tid; i < TT * NB; i += 256) s += nllbuf[i];
  red[tid] = s;
  __syncthreads();
  for (int st = 128; st >= 1; st >>= 1) {
    if (tid < st) red[tid] += red[tid + st];
    __syncthreads();
  }
  if (tid == 0) out[0] = red[0] * (1.0f / NB);
}

// ---------------- Host launcher ---------------------------------------------
extern "C" void kernel_launch(void* const* d_in, const int* in_sizes, int n_in,
                              void* d_out, int out_size, void* d_ws, size_t ws_size,
                              hipStream_t stream) {
  const float* features = (const float*)d_in[0];
  const int*   captions = (const int*)d_in[1];
  const float* W_embed  = (const float*)d_in[2];
  const float* W_proj   = (const float*)d_in[3];
  const float* b_proj   = (const float*)d_in[4];
  const float* Wx       = (const float*)d_in[5];
  const float* Wh       = (const float*)d_in[6];
  const float* Wattn    = (const float*)d_in[7];
  const float* b        = (const float*)d_in[8];
  const float* W_vocab  = (const float*)d_in[9];
  const float* b_vocab  = (const float*)d_in[10];
  float* out = (float*)d_out;

  char* ws = (char*)d_ws;
  size_t off = 0;
  bf16_t* WcatT  = (bf16_t*)(ws + off); off += (size_t)H4 * KC * 2;
  bf16_t* WvT    = (bf16_t*)(ws + off); off += (size_t)VV * HH * 2;
  bf16_t* x_bf   = (bf16_t*)(ws + off); off += (size_t)TT * NB * WW * 2;
  bf16_t* XHA    = (bf16_t*)(ws + off); off += (size_t)NB * KC * 2;
  float*  A_flat = (float*)(ws + off);  off += (size_t)NB * HH * 16 * 4;
  float*  hbuf   = (float*)(ws + off);  off += (size_t)NB * HH * 4;
  float*  cbuf   = (float*)(ws + off);  off += (size_t)NB * HH * 4;
  float*  zbuf   = (float*)(ws + off);  off += (size_t)NB * H4 * 4;
  bf16_t* hn     = (bf16_t*)(ws + off); off += (size_t)TT * NB * HH * 2;
  float*  nllbuf = (float*)(ws + off);  off += (size_t)TT * NB * 4;

  k_wcat  <<<(int)(((size_t)KC * H4 + 255) / 256), 256, 0, stream>>>(Wx, Wh, Wattn, WcatT);
  k_wvocab<<<(int)(((size_t)HH * VV + 255) / 256), 256, 0, stream>>>(W_vocab, WvT);
  k_embed <<<(int)(((size_t)TT * NB * WW + 255) / 256), 256, 0, stream>>>(captions, W_embed, x_bf);
  k_proj  <<<NB, 256, 0, stream>>>(features, W_proj, b_proj, A_flat, hbuf, cbuf);

  for (int t = 0; t < TT; t++) {
    k_attend<<<NB, 256, 0, stream>>>(t, hbuf, A_flat, x_bf, XHA);
    k_zgemm <<<dim3(H4 / 256, NB / 16), 256, 0, stream>>>(XHA, WcatT, b, zbuf);
    k_lstm  <<<(NB * HH + 255) / 256, 256, 0, stream>>>(t, zbuf, hbuf, cbuf, hn);
  }

  k_vocab<<<(TT * NB) / 16, 256, 0, stream>>>(hn, WvT, b_vocab, captions, nllbuf);
  k_loss <<<1, 256, 0, stream>>>(nllbuf, out);
}
